// ActorCritic_62620623175871
// MI455X (gfx1250) — matrix operand order repaired and checked by reading, not compiled, not run on hardware
//
#include <hip/hip_runtime.h>
#include <hip/hip_bf16.h>
#include <math.h>

typedef __attribute__((ext_vector_type(16))) _Float16 v16h;
typedef __attribute__((ext_vector_type(8)))  _Float16 v8h;
typedef __attribute__((ext_vector_type(4)))  _Float16 v4h;
typedef __attribute__((ext_vector_type(8)))  float    v8f;

#define B_  64
#define NT_ 1024
#define NU_ 32
#define H_  64
#define A_  8192

__device__ __forceinline__ float eluf(float x) {
    return x > 0.0f ? x : expm1f(x);
}

// Intra-wave LDS ordering: wait for this wave's outstanding DS ops; the
// "memory" clobber stops the compiler from hoisting later LDS reads above it.
__device__ __forceinline__ void wave_lds_fence() {
    asm volatile("s_wait_dscnt 0" ::: "memory");
}

__device__ __forceinline__ v16h cat8(v8h lo, v8h hi) {
    return __builtin_shufflevector(lo, hi, 0,1,2,3,4,5,6,7,8,9,10,11,12,13,14,15);
}

// ---------------------------------------------------------------------------
// Kernel 1: gathered action-input MLP -> scores[b][a]
// grid = (A/64, B), block = 128 (4 waves; each wave owns 16 action rows)
// ---------------------------------------------------------------------------
__global__ __launch_bounds__(128) void actor_scores_kernel(
    const float* __restrict__ task_embed,   // [B][NT][H]
    const float* __restrict__ usv_embed,    // [B][NU][H]
    const float* __restrict__ global_embed, // [B][H]
    const int*   __restrict__ task_ids,     // [A]
    const int*   __restrict__ usv_ids,      // [A]
    const float* __restrict__ W1,           // [192][64]
    const float* __restrict__ b1,           // [64]
    const float* __restrict__ W2,           // [64][32]
    const float* __restrict__ b2,           // [32]
    const float* __restrict__ W3,           // [32]
    const float* __restrict__ b3,           // [1]
    float* __restrict__ scores)             // [B][A]
{
    __shared__ __align__(16) _Float16 Wt1[64][192];   // W1 transposed: [n][k]
    __shared__ __align__(16) _Float16 Wt2[32][64];    // W2 transposed: [n][k]
    __shared__ __align__(16) _Float16 X  [4][16][192];// per-wave action tiles
    __shared__ __align__(16) _Float16 H1s[4][16][64]; // per-wave layer-1 act

    const int tid  = threadIdx.x;
    const int wave = tid >> 5;
    const int lane = tid & 31;
    const int b    = blockIdx.y;
    const int rowbase = blockIdx.x * 64 + wave * 16;

    // ---- stage transposed f16 weights (whole block) ----
    for (int i = tid; i < 192 * 64; i += 128) {
        int k = i >> 6, n = i & 63;
        Wt1[n][k] = (_Float16)W1[i];
    }
    for (int i = tid; i < 64 * 32; i += 128) {
        int k = i >> 5, n = i & 31;
        Wt2[n][k] = (_Float16)W2[i];
    }
    __syncthreads();

    // ---- stage 16 gathered rows: [t | u | g] as f16, 48 float4 per row ----
    for (int i = lane; i < 16 * 48; i += 32) {
        int r = i / 48, q = i % 48;
        int a = rowbase + r;
        const float4* src;
        if (q < 16) {
            src = (const float4*)(task_embed + ((size_t)b * NT_ + task_ids[a]) * H_) + q;
        } else if (q < 32) {
            src = (const float4*)(usv_embed + ((size_t)b * NU_ + usv_ids[a]) * H_) + (q - 16);
        } else {
            src = (const float4*)(global_embed + (size_t)b * H_) + (q - 32);
        }
        float4 v = *src;
        v4h h = { (_Float16)v.x, (_Float16)v.y, (_Float16)v.z, (_Float16)v.w };
        *(v4h*)&X[wave][r][q * 4] = h;
    }
    wave_lds_fence();

    const int row = lane & 15;   // A-row / B-column / C-column within fragment
    const int grp = lane >> 4;   // lane half-group

    // ---- layer 1: X(16x192) @ W1(192x64) via 6 K-steps x 4 N-tiles ----
    const v8f zacc = {0.f,0.f,0.f,0.f,0.f,0.f,0.f,0.f};
    v8f acc1[4] = {zacc, zacc, zacc, zacc};

    #pragma unroll
    for (int ks = 0; ks < 6; ++ks) {
        const _Float16* xr = &X[wave][row][0];
        // A 16x32 layout: lanes 0-15: K 0-7 (v0-3) & 16-23 (v4-7); lanes 16-31: +8
        v8h alo = *(const v8h*)&xr[ks * 32 + grp * 8];
        v8h ahi = *(const v8h*)&xr[ks * 32 + 16 + grp * 8];
        v16h afrag = cat8(alo, ahi);
        #pragma unroll
        for (int nt = 0; nt < 4; ++nt) {
            // B 32x16 layout: column = lane&15; lanes 0-15: K 0-7 & 16-23; lanes 16-31: +8
            const _Float16* wr = &Wt1[nt * 16 + row][0];
            v8h blo = *(const v8h*)&wr[ks * 32 + grp * 8];
            v8h bhi = *(const v8h*)&wr[ks * 32 + 16 + grp * 8];
            v16h bfrag = cat8(blo, bhi);
            acc1[nt] = __builtin_amdgcn_wmma_f32_16x16x32_f16(
                false, afrag, false, bfrag, (short)0, acc1[nt], false, false);
        }
    }

    // ---- epilogue 1: bias + ELU -> H1s (16x64 f16) ----
    #pragma unroll
    for (int nt = 0; nt < 4; ++nt) {
        float bias = b1[nt * 16 + row];
        #pragma unroll
        for (int r = 0; r < 8; ++r) {
            float x = acc1[nt][r] + bias;          // C layout: M=r+8*grp, N=nt*16+row
            H1s[wave][r + 8 * grp][nt * 16 + row] = (_Float16)eluf(x);
        }
    }
    wave_lds_fence();

    // ---- layer 2: H1(16x64) @ W2(64x32) via 2 K-steps x 2 N-tiles ----
    v8f acc2[2] = {zacc, zacc};
    #pragma unroll
    for (int ks = 0; ks < 2; ++ks) {
        const _Float16* hr = &H1s[wave][row][0];
        v8h alo = *(const v8h*)&hr[ks * 32 + grp * 8];
        v8h ahi = *(const v8h*)&hr[ks * 32 + 16 + grp * 8];
        v16h afrag = cat8(alo, ahi);
        #pragma unroll
        for (int nt = 0; nt < 2; ++nt) {
            const _Float16* wr = &Wt2[nt * 16 + row][0];
            v8h blo = *(const v8h*)&wr[ks * 32 + grp * 8];
            v8h bhi = *(const v8h*)&wr[ks * 32 + 16 + grp * 8];
            v16h bfrag = cat8(blo, bhi);
            acc2[nt] = __builtin_amdgcn_wmma_f32_16x16x32_f16(
                false, afrag, false, bfrag, (short)0, acc2[nt], false, false);
        }
    }

    // ---- layer 3: elu(h2 + b2) . W3 + b3, reduced across the 16-lane group ----
    float bb0 = b2[row],      bb1 = b2[16 + row];
    float w30 = W3[row],      w31 = W3[16 + row];
    float b3v = b3[0];
    #pragma unroll
    for (int r = 0; r < 8; ++r) {
        float s = eluf(acc2[0][r] + bb0) * w30 + eluf(acc2[1][r] + bb1) * w31;
        s += __shfl_xor(s, 1, 32);
        s += __shfl_xor(s, 2, 32);
        s += __shfl_xor(s, 4, 32);
        s += __shfl_xor(s, 8, 32);   // all 16 lanes of each half now hold row sum
        if (row == 0) {              // lane 0 -> rows 0..7, lane 16 -> rows 8..15
            scores[(size_t)b * A_ + rowbase + r + 8 * grp] = s + b3v;
        }
    }
}

// ---------------------------------------------------------------------------
// Kernel 2: per-batch softmax over A=8192 scores
// ---------------------------------------------------------------------------
__global__ __launch_bounds__(256) void softmax_kernel(
    const float* __restrict__ scores, float* __restrict__ probs)
{
    __shared__ float red[256];
    const int b = blockIdx.x, tid = threadIdx.x;
    const float* s = scores + (size_t)b * A_;

    float m = -INFINITY;
    for (int i = tid; i < A_; i += 256) m = fmaxf(m, s[i]);
    red[tid] = m; __syncthreads();
    for (int w = 128; w > 0; w >>= 1) {
        if (tid < w) red[tid] = fmaxf(red[tid], red[tid + w]);
        __syncthreads();
    }
    const float bm = red[0];
    __syncthreads();

    float sum = 0.0f;
    for (int i = tid; i < A_; i += 256) sum += expf(s[i] - bm);
    red[tid] = sum; __syncthreads();
    for (int w = 128; w > 0; w >>= 1) {
        if (tid < w) red[tid] += red[tid + w];
        __syncthreads();
    }
    const float inv = 1.0f / red[0];

    for (int i = tid; i < A_; i += 256)
        probs[(size_t)b * A_ + i] = expf(s[i] - bm) * inv;
}

// ---------------------------------------------------------------------------
// Kernel 3: critic MLP on global_embed (64 rows) -> state_value[64]
// ---------------------------------------------------------------------------
__global__ __launch_bounds__(64) void critic_kernel(
    const float* __restrict__ global_embed,
    const float* __restrict__ Wc1, const float* __restrict__ bc1,
    const float* __restrict__ Wc2, const float* __restrict__ bc2,
    const float* __restrict__ Wc3, const float* __restrict__ bc3,
    float* __restrict__ out)
{
    const int t = threadIdx.x;   // one thread per batch row
    float g[64];
    #pragma unroll 8
    for (int k = 0; k < 64; ++k) g[k] = global_embed[t * 64 + k];

    float h1[64];
    for (int j = 0; j < 64; ++j) {
        float s = bc1[j];
        #pragma unroll 8
        for (int k = 0; k < 64; ++k) s += g[k] * Wc1[k * 64 + j];
        h1[j] = eluf(s);
    }
    float h2[32];
    for (int j = 0; j < 32; ++j) {
        float s = bc2[j];
        #pragma unroll 8
        for (int k = 0; k < 64; ++k) s += h1[k] * Wc2[k * 32 + j];
        h2[j] = eluf(s);
    }
    float v = bc3[0];
    #pragma unroll 8
    for (int j = 0; j < 32; ++j) v += h2[j] * Wc3[j];
    out[t] = v;
}

// ---------------------------------------------------------------------------
extern "C" void kernel_launch(void* const* d_in, const int* in_sizes, int n_in,
                              void* d_out, int out_size, void* d_ws, size_t ws_size,
                              hipStream_t stream) {
    const float* task_embed   = (const float*)d_in[0];
    const float* usv_embed    = (const float*)d_in[1];
    const float* global_embed = (const float*)d_in[2];
    const int*   task_ids     = (const int*)  d_in[3];
    const int*   usv_ids      = (const int*)  d_in[4];
    const float* W1  = (const float*)d_in[5];
    const float* b1  = (const float*)d_in[6];
    const float* W2  = (const float*)d_in[7];
    const float* b2  = (const float*)d_in[8];
    const float* W3  = (const float*)d_in[9];
    const float* b3  = (const float*)d_in[10];
    const float* Wc1 = (const float*)d_in[11];
    const float* bc1 = (const float*)d_in[12];
    const float* Wc2 = (const float*)d_in[13];
    const float* bc2 = (const float*)d_in[14];
    const float* Wc3 = (const float*)d_in[15];
    const float* bc3 = (const float*)d_in[16];

    float* probs  = (float*)d_out;                       // [B][A]
    float* value  = (float*)d_out + (size_t)B_ * A_;     // [B]
    float* scores = (float*)d_ws;                        // [B][A] scratch (2 MB)

    dim3 grid(A_ / 64, B_);
    actor_scores_kernel<<<grid, 128, 0, stream>>>(
        task_embed, usv_embed, global_embed, task_ids, usv_ids,
        W1, b1, W2, b2, W3, b3, scores);
    softmax_kernel<<<B_, 256, 0, stream>>>(scores, probs);
    critic_kernel<<<1, B_, 0, stream>>>(global_embed, Wc1, bc1, Wc2, bc2, Wc3, bc3, value);
}
